// NaMultiHeadAttention_9826885173351
// MI455X (gfx1250) — matrix operand order repaired and checked by reading, not compiled, not run on hardware
//
#include <hip/hip_runtime.h>
#include <hip/hip_bf16.h>

typedef __attribute__((ext_vector_type(16))) _Float16 v16h;
typedef __attribute__((ext_vector_type(8)))  _Float16 v8h;
typedef __attribute__((ext_vector_type(8)))  float    v8f;
typedef __attribute__((ext_vector_type(4)))  float    v4f;

static constexpr int Bn = 4, Nn = 4096, En = 256, Hn = 4, Dn = 64;
static constexpr int Mrows = Bn * Nn; // 16384

__device__ __forceinline__ v8f wmma_f16(v16h a, v16h b, v8f c) {
  // D = A(16x32 f16) * B(32x16 f16) + C(16x16 f32)
  return __builtin_amdgcn_wmma_f32_16x16x32_f16(false, a, false, b, (short)0, c,
                                                false, false);
}

// ---- operand loaders -------------------------------------------------------
// A operand (16x32, f16): lane = row (lane&15); K chunks {koff..koff+7, koff+16..koff+23},
// koff = 0 (lanes 0-15) or 8 (lanes 16-31), per ISA 16-bit A-matrix table.
__device__ __forceinline__ v16h load_a_f16(const _Float16* base, int ld) {
  const int lane = threadIdx.x & 31;
  const int row  = lane & 15;
  const int koff = (lane & 16) ? 8 : 0;
  const _Float16* p = base + row * ld + koff;
  v8h lo = *(const v8h*)(p);
  v8h hi = *(const v8h*)(p + 16);
  v16h r;
#pragma unroll
  for (int i = 0; i < 8; ++i) { r[i] = lo[i]; r[8 + i] = hi[i]; }
  return r;
}

__device__ __forceinline__ v16h load_a_f32cvt(const float* base, int ld) {
  const int lane = threadIdx.x & 31;
  const int row  = lane & 15;
  const int koff = (lane & 16) ? 8 : 0;
  const float* p = base + row * ld + koff;
  v16h r;
#pragma unroll
  for (int i = 0; i < 8; ++i) { r[i] = (_Float16)p[i]; r[8 + i] = (_Float16)p[16 + i]; }
  return r;
}

// B operand (32x16, f16) read from row-major B^T storage [N x ld]:
// lane = output column (lane&15); K chunks {koff..koff+7, koff+16..koff+23},
// koff = 0 (lanes 0-15) or 8 (lanes 16-31), matching the A-operand K order.
__device__ __forceinline__ v16h load_b_f16(const _Float16* base, int ld) {
  const int lane = threadIdx.x & 31;
  const int col  = lane & 15;
  const int koff = (lane & 16) ? 8 : 0;
  const _Float16* p = base + col * ld + koff;
  v8h lo = *(const v8h*)(p);
  v8h hi = *(const v8h*)(p + 16);
  v16h r;
#pragma unroll
  for (int i = 0; i < 8; ++i) { r[i] = lo[i]; r[8 + i] = hi[i]; }
  return r;
}

__device__ __forceinline__ v16h load_b_f32cvt(const float* base, int ld) {
  const int lane = threadIdx.x & 31;
  const int col  = lane & 15;
  const int koff = (lane & 16) ? 8 : 0;
  const float* p = base + col * ld + koff;
  v16h r;
#pragma unroll
  for (int i = 0; i < 8; ++i) { r[i] = (_Float16)p[i]; r[8 + i] = (_Float16)p[16 + i]; }
  return r;
}

// ---- kernel 0: q passthrough ----------------------------------------------
__global__ void copy_q_kernel(const float* __restrict__ src, float* __restrict__ dst, int n4) {
  int i = blockIdx.x * blockDim.x + threadIdx.x;
  if (i < n4) ((v4f*)dst)[i] = ((const v4f*)src)[i];
}

// ---- kernel 1: fused QKV projection (X @ W^T + b), f32 -> f16 --------------
// z = 0: Q -> Qp [B*N, E];  z = 1: K -> Kp [B*N, E];  z = 2: V -> Vt [B,H,D,N] (transposed)
__global__ void qkv_proj_kernel(const float* __restrict__ q, const float* __restrict__ k,
                                const float* __restrict__ v,
                                const float* __restrict__ Wq, const float* __restrict__ bq,
                                const float* __restrict__ Wk, const float* __restrict__ bk,
                                const float* __restrict__ Wv, const float* __restrict__ bv,
                                _Float16* __restrict__ Qp, _Float16* __restrict__ Kp,
                                _Float16* __restrict__ Vt) {
  const int z = blockIdx.z;
  const float* X    = (z == 0) ? q  : (z == 1) ? k  : v;
  const float* W    = (z == 0) ? Wq : (z == 1) ? Wk : Wv;
  const float* bias = (z == 0) ? bq : (z == 1) ? bk : bv;
  const int mTile = blockIdx.x * 16;
  const int nTile = blockIdx.y * 64;

  v8f c[4] = {};
  for (int kk = 0; kk < En; kk += 32) {
    v16h a = load_a_f32cvt(X + (size_t)mTile * En + kk, En);
#pragma unroll
    for (int t = 0; t < 4; ++t)
      c[t] = wmma_f16(a, load_b_f32cvt(W + (size_t)(nTile + t * 16) * En + kk, En), c[t]);
  }

  const int lane   = threadIdx.x & 31;
  const int colL   = lane & 15;
  const int rowOff = (lane >> 4) * 8;
#pragma unroll
  for (int t = 0; t < 4; ++t) {
    const int col = nTile + t * 16 + colL;
    const float bb = bias[col];
#pragma unroll
    for (int j = 0; j < 8; ++j) {
      const int row = mTile + rowOff + j;
      const float val = c[t][j] + bb;
      if (z < 2) {
        _Float16* Y = (z == 0) ? Qp : Kp;
        Y[(size_t)row * En + col] = (_Float16)val;
      } else {
        const int bidx = row >> 12, nrow = row & (Nn - 1);
        const int hh = col >> 6, d = col & (Dn - 1);
        Vt[(((size_t)(bidx * Hn + hh) * Dn) + d) * Nn + nrow] = (_Float16)val;
      }
    }
  }
}

// ---- kernel 2: flash attention, one wave per 32-query tile -----------------
// Per 32-key chunk: 8 WMMA for S (K tiles reused by both row-blocks), 8 WMMA for P.V
// (V tiles reused). Softmax: S goes through LDS (C-layout store, A-layout f32 reload)
// so row reductions are within-lane + one shfl_xor(16); exp output converts directly
// into the f16 A operand for P.V (no second LDS round trip).
__global__ void attn_kernel(const _Float16* __restrict__ Qp, const _Float16* __restrict__ Kp,
                            const _Float16* __restrict__ Vt, _Float16* __restrict__ Ao) {
  __shared__ __align__(16) float smemS[2 * 16 * 32];  // raw S per row-block
  const int lane  = threadIdx.x & 31;
  const int qTile = blockIdx.x * 32;
  const int b = blockIdx.y >> 2;
  const int h = blockIdx.y & 3;

  const _Float16* Qb = Qp + ((size_t)(b * Nn + qTile)) * En + h * Dn;
  const _Float16* Kb = Kp + (size_t)b * Nn * En + h * Dn;
  const _Float16* Vb = Vt + (size_t)(b * Hn + h) * Dn * Nn;

  v16h aq[2][2];
#pragma unroll
  for (int rb = 0; rb < 2; ++rb) {
    aq[rb][0] = load_a_f16(Qb + (size_t)(rb * 16) * En, En);       // d 0..31
    aq[rb][1] = load_a_f16(Qb + (size_t)(rb * 16) * En + 32, En);  // d 32..63
  }

  v8f o[2][4] = {};            // two 16x64 f32 accumulators
  float mrun[2], lrun[2];      // running stats, row = lane&15 (replicated in half-waves)
#pragma unroll
  for (int rb = 0; rb < 2; ++rb) { mrun[rb] = -1e30f; lrun[rb] = 0.0f; }

  const int rowSel = ((lane & 16) >> 1);   // +8 for upper half-wave (C-layout row offset)
  const int rowA   = lane & 15;            // A-layout row for this lane
  const int koffA  = (lane & 16) ? 8 : 0;  // A-layout K sub-chunk

  for (int kc = 0; kc < Nn; kc += 32) {
    // ---- S = Q K^T : 4 K B-tiles shared by both row-blocks (8 WMMA) ----
    v16h bk0 = load_b_f16(Kb + (size_t)(kc)      * En,      En);
    v16h bk1 = load_b_f16(Kb + (size_t)(kc)      * En + 32, En);
    v16h bk2 = load_b_f16(Kb + (size_t)(kc + 16) * En,      En);
    v16h bk3 = load_b_f16(Kb + (size_t)(kc + 16) * En + 32, En);
#pragma unroll
    for (int rb = 0; rb < 2; ++rb) {
      v8f s0 = {}, s1 = {};
      s0 = wmma_f16(aq[rb][0], bk0, s0);
      s0 = wmma_f16(aq[rb][1], bk1, s0);
      s1 = wmma_f16(aq[rb][0], bk2, s1);
      s1 = wmma_f16(aq[rb][1], bk3, s1);
      float* sm = smemS + rb * (16 * 32);
#pragma unroll
      for (int j = 0; j < 8; ++j) {            // C-layout -> row-major f32 in LDS
        const int r = j + rowSel;
        sm[r * 32 + (lane & 15)]      = s0[j];
        sm[r * 32 + 16 + (lane & 15)] = s1[j];
      }
    }
    __syncthreads();

    // ---- V B-tiles shared by both row-blocks ----
    v16h bv0 = load_b_f16(Vb + (size_t)(0)  * Nn + kc, Nn);
    v16h bv1 = load_b_f16(Vb + (size_t)(16) * Nn + kc, Nn);
    v16h bv2 = load_b_f16(Vb + (size_t)(32) * Nn + kc, Nn);
    v16h bv3 = load_b_f16(Vb + (size_t)(48) * Nn + kc, Nn);

#pragma unroll
    for (int rb = 0; rb < 2; ++rb) {
      // reload S already in A-operand order (16 of 32 K per lane)
      const float* p = smemS + rb * (16 * 32) + rowA * 32 + koffA;
      float vals[16];
#pragma unroll
      for (int i = 0; i < 8; ++i) { vals[i] = p[i]; vals[8 + i] = p[16 + i]; }

      // row max: within-lane tree + combine the two half-lanes of this row
      float mx = vals[0];
#pragma unroll
      for (int i = 1; i < 16; ++i) mx = fmaxf(mx, vals[i]);
      mx = fmaxf(mx, __shfl_xor(mx, 16, 32));
      const float newm = fmaxf(mrun[rb], mx);
      const float corr = __expf(mrun[rb] - newm);
      mrun[rb] = newm;

      // exp -> f16 A operand directly; within-lane row sum
      v16h pa;
      float rs = 0.0f;
#pragma unroll
      for (int i = 0; i < 16; ++i) {
        const float pv = __expf(vals[i] - newm);
        rs += pv;
        pa[i] = (_Float16)pv;
      }
      rs += __shfl_xor(rs, 16, 32);
      lrun[rb] = lrun[rb] * corr + rs;

      // rescale O (broadcast corr from row-owner lane into C layout)
#pragma unroll
      for (int j = 0; j < 8; ++j) {
        const float cc = __shfl(corr, j + rowSel, 32);
        o[rb][0][j] *= cc; o[rb][1][j] *= cc; o[rb][2][j] *= cc; o[rb][3][j] *= cc;
      }

      o[rb][0] = wmma_f16(pa, bv0, o[rb][0]);
      o[rb][1] = wmma_f16(pa, bv1, o[rb][1]);
      o[rb][2] = wmma_f16(pa, bv2, o[rb][2]);
      o[rb][3] = wmma_f16(pa, bv3, o[rb][3]);
    }
    __syncthreads();
  }

  // normalize (fold the reference's post-softmax /sqrt(E) = /16) and store f16
  const int colL   = lane & 15;
  const int rowOff = (lane >> 4) * 8;
#pragma unroll
  for (int rb = 0; rb < 2; ++rb) {
    const float inv = 1.0f / (lrun[rb] * 16.0f);
#pragma unroll
    for (int j = 0; j < 8; ++j) {
      const float invC = __shfl(inv, j + rowSel, 32);
      const int row = qTile + rb * 16 + rowOff + j;
      _Float16* dst = Ao + ((size_t)(b * Nn + row)) * En + h * Dn;
#pragma unroll
      for (int t = 0; t < 4; ++t)
        dst[t * 16 + colL] = (_Float16)(o[rb][t][j] * invC);
    }
  }
}

// ---- kernel 3: output projection (Ao @ Wo^T + bo), writes both out copies --
__global__ void out_proj_kernel(const _Float16* __restrict__ Ao, const float* __restrict__ Wo,
                                const float* __restrict__ bo, float* __restrict__ out1,
                                float* __restrict__ out2) {
  const int mTile = blockIdx.x * 16;
  const int nTile = blockIdx.y * 64;
  v8f c[4] = {};
  for (int kk = 0; kk < En; kk += 32) {
    v16h a = load_a_f16(Ao + (size_t)mTile * En + kk, En);
#pragma unroll
    for (int t = 0; t < 4; ++t)
      c[t] = wmma_f16(a, load_b_f32cvt(Wo + (size_t)(nTile + t * 16) * En + kk, En), c[t]);
  }
  const int lane   = threadIdx.x & 31;
  const int colL   = lane & 15;
  const int rowOff = (lane >> 4) * 8;
#pragma unroll
  for (int t = 0; t < 4; ++t) {
    const int col = nTile + t * 16 + colL;
    const float bb = bo[col];
#pragma unroll
    for (int j = 0; j < 8; ++j) {
      const int row = mTile + rowOff + j;
      const float val = c[t][j] + bb;
      out1[(size_t)row * En + col] = val;
      out2[(size_t)row * En + col] = val;
    }
  }
}

extern "C" void kernel_launch(void* const* d_in, const int* in_sizes, int n_in,
                              void* d_out, int out_size, void* d_ws, size_t ws_size,
                              hipStream_t stream) {
  (void)in_sizes; (void)n_in; (void)out_size; (void)ws_size;
  const float* q  = (const float*)d_in[0];
  const float* k  = (const float*)d_in[1];
  const float* v  = (const float*)d_in[2];
  const float* Wq = (const float*)d_in[3];
  const float* bq = (const float*)d_in[4];
  const float* Wk = (const float*)d_in[5];
  const float* bk = (const float*)d_in[6];
  const float* Wv = (const float*)d_in[7];
  const float* bv = (const float*)d_in[8];
  const float* Wo = (const float*)d_in[9];
  const float* bo = (const float*)d_in[10];

  float* out = (float*)d_out;
  const size_t seg = (size_t)Bn * Nn * En;  // 4,194,304 elements per output segment

  _Float16* ws = (_Float16*)d_ws;           // 4 x 8MB f16 buffers = 32 MB
  _Float16* Qp = ws;
  _Float16* Kp = ws + seg;
  _Float16* Vt = ws + 2 * seg;
  _Float16* Ao = ws + 3 * seg;

  // segment 0: q passthrough (vectorized f32x4 copy)
  copy_q_kernel<<<dim3((unsigned)(seg / (256 * 4))), dim3(256), 0, stream>>>(q, out, (int)(seg / 4));

  // QKV projections -> f16 workspace (V transposed per head)
  qkv_proj_kernel<<<dim3(Mrows / 16, En / 64, 3), dim3(32), 0, stream>>>(
      q, k, v, Wq, bq, Wk, bk, Wv, bv, Qp, Kp, Vt);

  // flash attention (unscaled logits; /sqrt(E) folded into normalization)
  attn_kernel<<<dim3(Nn / 32, Bn * Hn), dim3(32), 0, stream>>>(Qp, Kp, Vt, Ao);

  // output projection -> segments 1 and 2
  out_proj_kernel<<<dim3(Mrows / 16, En / 64), dim3(32), 0, stream>>>(
      Ao, Wo, bo, out + seg, out + 2 * seg);
}
